// Decoder_48687749267883
// MI455X (gfx1250) — matrix operand order repaired and checked by reading, not compiled, not run on hardware
//
#include <hip/hip_runtime.h>
#include <hip/hip_bf16.h>
#include <math.h>

// Problem constants (from reference)
#define B_   512
#define D_   512
#define SEQ_ 50
#define NH_  8
#define HD_  64
#define FF_  2048
#define HI_  497   // max(512-15,16)

typedef __attribute__((ext_vector_type(16))) _Float16 v16h;
typedef __attribute__((ext_vector_type(8)))  _Float16 v8h;
typedef __attribute__((ext_vector_type(8)))  float    v8f;
typedef __attribute__((ext_vector_type(4)))  float    v4f;

// ---------------- wave32 reductions ----------------
__device__ __forceinline__ float wred_sum(float v) {
#pragma unroll
  for (int o = 16; o >= 1; o >>= 1) v += __shfl_xor(v, o, 32);
  return v;
}
__device__ __forceinline__ float wred_max(float v) {
#pragma unroll
  for (int o = 16; o >= 1; o >>= 1) v = fmaxf(v, __shfl_xor(v, o, 32));
  return v;
}

// ---------------- WMMA GEMM: C[M,N] = act(A[M,K] @ W[K,N] + bias + extra) ----
// A: fp32 (or f16 if AH) row-major. WT: f16, W transposed -> [N][K] row-major.
// Wave tile: (TWOM ? 32 : 16) rows x 64 cols -> (TWOM ? 8 : 4) WMMAs per k-step,
// B fragments shared across the two M sub-tiles.
// K % 32 == 0; M % (TWOM?32:16) == 0; N arbitrary iff !FULLN (guarded).
// EPI: 0=none, 1=exact GELU, 2=ELU.  OUTH: write f16 output (KV cache).
template<int EPI, int OUTH, int AH, int FULLN, int TWOM>
__global__ __launch_bounds__(256) void gemm_wmma_kernel(
    const float* __restrict__ A, const _Float16* __restrict__ Ah,
    const _Float16* __restrict__ WT,
    const float* __restrict__ bias, const float* __restrict__ extra,
    float* __restrict__ Cf, _Float16* __restrict__ Ch,
    int M, int N, int K)
{
  const int lane = threadIdx.x & 31;
  const int wv   = blockIdx.x * 8 + (threadIdx.x >> 5);
  const int mTiles  = M >> (TWOM ? 5 : 4);
  const int nGroups = (N + 63) >> 6;
  if (wv >= mTiles * nGroups) return;
  const int mt = wv % mTiles;          // consecutive waves share weights (L2 reuse)
  const int ng = wv / mTiles;
  const int hf = lane >> 4;            // lane half (ISA A/B fragment layout)
  const int r  = lane & 15;
  const int n0 = (ng << 6) + r;
  const int row0 = (mt << (TWOM ? 5 : 4)) + r;

  v8f acc[TWOM ? 8 : 4] = {};

  for (int k0 = 0; k0 < K; k0 += 32) {
    // ---- A fragments (16x32 f16): lane hf/r holds row r,
    //      K = k0+hf*8+{0..7} (elems 0..7) and k0+16+hf*8+{0..7} (elems 8..15)
    v16h a0, a1;
    if (AH) {
      const _Float16* p = Ah + (size_t)row0 * K + k0 + hf * 8;
      v8h x0 = *(const v8h*)p;
      v8h x1 = *(const v8h*)(p + 16);
#pragma unroll
      for (int i = 0; i < 8; i++) { a0[i] = x0[i]; a0[8 + i] = x1[i]; }
      if (TWOM) {
        const _Float16* p1 = p + (size_t)16 * K;
        v8h y0 = *(const v8h*)p1;
        v8h y1 = *(const v8h*)(p1 + 16);
#pragma unroll
        for (int i = 0; i < 8; i++) { a1[i] = y0[i]; a1[8 + i] = y1[i]; }
      }
    } else {
      const float* p = A + (size_t)row0 * K + k0 + hf * 8;
      v4f f0 = *(const v4f*)p;
      v4f f1 = *(const v4f*)(p + 4);
      v4f f2 = *(const v4f*)(p + 16);
      v4f f3 = *(const v4f*)(p + 20);
#pragma unroll
      for (int i = 0; i < 4; i++) {
        a0[i]      = (_Float16)f0[i];
        a0[4 + i]  = (_Float16)f1[i];
        a0[8 + i]  = (_Float16)f2[i];
        a0[12 + i] = (_Float16)f3[i];
      }
      if (TWOM) {
        const float* q = p + (size_t)16 * K;
        v4f g0 = *(const v4f*)q;
        v4f g1 = *(const v4f*)(q + 4);
        v4f g2 = *(const v4f*)(q + 16);
        v4f g3 = *(const v4f*)(q + 20);
#pragma unroll
        for (int i = 0; i < 4; i++) {
          a1[i]      = (_Float16)g0[i];
          a1[4 + i]  = (_Float16)g1[i];
          a1[8 + i]  = (_Float16)g2[i];
          a1[12 + i] = (_Float16)g3[i];
        }
      }
    }
    // ---- 4 B fragments (32x16 f16): lane holds column n, K = k0+hf*8+{0..7} and k0+16+hf*8+{0..7}
    v16h b[4];
#pragma unroll
    for (int j = 0; j < 4; j++) {
      const int n = n0 + j * 16;
      if (FULLN || n < N) {
        const _Float16* bp = WT + (size_t)n * K + k0 + hf * 8;
        v8h lo = *(const v8h*)bp;
        v8h hi = *(const v8h*)(bp + 16);
#pragma unroll
        for (int i = 0; i < 8; i++) { b[j][i] = lo[i]; b[j][8 + i] = hi[i]; }
      } else {
#pragma unroll
        for (int i = 0; i < 16; i++) b[j][i] = (_Float16)0.f;
      }
    }
#pragma unroll
    for (int j = 0; j < 4; j++) {
      acc[j] = __builtin_amdgcn_wmma_f32_16x16x32_f16(
          false, a0, false, b[j], (short)0, acc[j], false, false);
      if (TWOM)
        acc[4 + j] = __builtin_amdgcn_wmma_f32_16x16x32_f16(
            false, a1, false, b[j], (short)0, acc[4 + j], false, false);
    }
  }

  // ---- epilogue: C layout VGPR i -> row = tileBase + m*16 + 8*hf + i, col n0+j*16
#pragma unroll
  for (int j = 0; j < 4; j++) {
    const int n = n0 + j * 16;
    if (!FULLN && n >= N) continue;
    const float add = (bias ? bias[n] : 0.f) + (extra ? extra[n] : 0.f);
#pragma unroll
    for (int m = 0; m < (TWOM ? 2 : 1); m++) {
#pragma unroll
      for (int i = 0; i < 8; i++) {
        const int row = (mt << (TWOM ? 5 : 4)) + m * 16 + (hf << 3) + i;
        float v = acc[m * 4 + j][i] + add;
        if (EPI == 1)      v = 0.5f * v * (1.0f + erff(v * 0.70710678118654752f));
        else if (EPI == 2) v = (v > 0.f) ? v : (expf(v) - 1.f);
        if (OUTH) Ch[(size_t)row * N + n] = (_Float16)v;
        else      Cf[(size_t)row * N + n] = v;
      }
    }
  }
}

// ---------------- weight convert+transpose: W[K][N] fp32 -> WT[N][K] f16 -----
__global__ __launch_bounds__(256) void convT_kernel(
    const float* __restrict__ W, _Float16* __restrict__ WT, int K, int N)
{
  int idx = blockIdx.x * 256 + threadIdx.x;
  if (idx >= K * N) return;
  int k = idx / N, n = idx - k * N;
  WT[(size_t)n * K + k] = (_Float16)W[idx];
}

// ---------------- flat fp32 -> f16 copy (activation pre-convert) ------------
__global__ __launch_bounds__(256) void convA_kernel(
    const float* __restrict__ A, _Float16* __restrict__ Ahalf, int n)
{
  int i = blockIdx.x * 256 + threadIdx.x;
  if (i < n) Ahalf[i] = (_Float16)A[i];
}

// ---------------- embedding: e = [x0@Wb+bb | tn@Wt+bt | prev@Wp+bp] ---------
__global__ __launch_bounds__(256) void embed_kernel(
    const float* __restrict__ x0, const float* __restrict__ times,
    const float* __restrict__ prev,
    const float* __restrict__ Wb, const float* __restrict__ bb,
    const float* __restrict__ Wt, const float* __restrict__ bt,
    const float* __restrict__ Wp, const float* __restrict__ bp,
    float* __restrict__ e, int t)
{
  int idx = blockIdx.x * 256 + threadIdx.x;
  if (idx >= B_ * 3 * D_) return;
  int b = idx / (3 * D_), c = idx - b * (3 * D_);
  float v;
  if (c < D_) {
    float s = bb[c];
    const float* xr = x0 + b * 29;
#pragma unroll
    for (int j = 0; j < 29; j++) s += xr[j] * Wb[j * D_ + c];
    v = s;
  } else if (c < 2 * D_) {
    int n = c - D_;
    float tn = (times[b * SEQ_ + t] - 70.0f) * 0.1f;
    v = tn * Wt[n] + bt[n];
  } else {
    int n = c - 2 * D_;
    v = prev[b * 2 + 0] * Wp[n] + prev[b * 2 + 1] * Wp[D_ + n] + bp[n];
  }
  e[(size_t)b * (3 * D_) + c] = v;
}

// ---------------- LayerNorm: out = LN(x + res) * g + b (eps=1e-5) -----------
__global__ __launch_bounds__(256) void ln_kernel(
    const float* __restrict__ x, const float* __restrict__ res,
    const float* __restrict__ g, const float* __restrict__ bta,
    float* __restrict__ out)
{
  const int lane = threadIdx.x & 31;
  const int row  = blockIdx.x * 8 + (threadIdx.x >> 5);
  if (row >= B_) return;
  const float* xr = x + (size_t)row * D_;
  const float* rr = res + (size_t)row * D_;
  float v[16]; float s = 0.f;
#pragma unroll
  for (int i = 0; i < 16; i++) { int c = lane + i * 32; float t = xr[c] + rr[c]; v[i] = t; s += t; }
  s = wred_sum(s);
  const float mean = s * (1.f / 512.f);
  float var = 0.f;
#pragma unroll
  for (int i = 0; i < 16; i++) { float d = v[i] - mean; var += d * d; }
  var = wred_sum(var) * (1.f / 512.f);
  const float inv = 1.f / sqrtf(var + 1e-5f);
  float* orow = out + (size_t)row * D_;
#pragma unroll
  for (int i = 0; i < 16; i++) { int c = lane + i * 32; orow[c] = (v[i] - mean) * inv * g[c] + bta[c]; }
}

// ---------------- cross-attention: one wave per (b, head), 50 keys ----------
__global__ __launch_bounds__(256) void attn_kernel(
    const float* __restrict__ q, const _Float16* __restrict__ Kh,
    const _Float16* __restrict__ Vh, float* __restrict__ out)
{
  __shared__ float sc[8][56];
  const int lane = threadIdx.x & 31;
  const int w  = threadIdx.x >> 5;
  const int gw = blockIdx.x * 8 + w;          // grid sized exactly B_*NH_ waves
  const int b = gw >> 3, h = gw & 7;
  const float* qp = q + (size_t)b * D_ + h * HD_;
  const float q0 = qp[lane * 2], q1 = qp[lane * 2 + 1];
  const _Float16* Kb = Kh + (size_t)b * SEQ_ * D_ + h * HD_;
  for (int s = 0; s < SEQ_; s++) {
    const _Float16* kp = Kb + (size_t)s * D_;
    float p = q0 * (float)kp[lane * 2] + q1 * (float)kp[lane * 2 + 1];
    p = wred_sum(p);
    if (lane == 0) sc[w][s] = p * 0.125f;     // 1/sqrt(64)
  }
  __syncthreads();
  float m = -1e30f;
  for (int s = lane; s < SEQ_; s += 32) m = fmaxf(m, sc[w][s]);
  m = wred_max(m);
  float sum = 0.f;
  for (int s = lane; s < SEQ_; s += 32) { float e = expf(sc[w][s] - m); sc[w][s] = e; sum += e; }
  sum = wred_sum(sum);
  __syncthreads();
  const float inv = 1.f / sum;
  const _Float16* Vb = Vh + (size_t)b * SEQ_ * D_ + h * HD_;
  float a0 = 0.f, a1 = 0.f;
  for (int s = 0; s < SEQ_; s++) {
    float p = sc[w][s];
    const _Float16* vp = Vb + (size_t)s * D_;
    a0 += p * (float)vp[lane * 2];
    a1 += p * (float)vp[lane * 2 + 1];
  }
  out[(size_t)b * D_ + h * HD_ + lane * 2]     = a0 * inv;
  out[(size_t)b * D_ + h * HD_ + lane * 2 + 1] = a1 * inv;
}

// ---------------- hazard head tail: h = elu_act @ W2 + b2, clips, outputs ---
__global__ __launch_bounds__(256) void haz2_kernel(
    const float* __restrict__ hb, const float* __restrict__ W2,
    const float* __restrict__ b2, float* __restrict__ prev,
    float* __restrict__ out, int t)
{
  const int lane = threadIdx.x & 31;
  const int b = blockIdx.x * 8 + (threadIdx.x >> 5);
  if (b >= B_) return;
  float s = 0.f;
  for (int j = lane; j < HI_; j += 32) s += hb[(size_t)b * D_ + j] * W2[j];
  s = wred_sum(s) + b2[0];
  float lg = fminf(fmaxf(s, -10.f), 10.f);
  float hz = fminf(fmaxf(expf(lg), 1e-8f), 50.f);
  float sv = fminf(fmaxf(expf(-hz), 1e-8f), 1.f - 1e-8f);
  if (lane == 0) {
    out[(size_t)b * SEQ_ + t] = sv;                        // S  [B,SEQ]
    out[(size_t)B_ * SEQ_ + (size_t)b * SEQ_ + t] = lg;    // LG [B,SEQ]
    prev[b * 2 + 0] = lg;
    prev[b * 2 + 1] = sv;
  }
}

__global__ void zero_prev_kernel(float* prev) {
  int i = blockIdx.x * blockDim.x + threadIdx.x;
  if (i < B_ * 2) prev[i] = 0.f;
}

// ============================ host driver ===================================
extern "C" void kernel_launch(void* const* d_in, const int* in_sizes, int n_in,
                              void* d_out, int out_size, void* d_ws, size_t ws_size,
                              hipStream_t stream)
{
  (void)in_sizes; (void)out_size;
  if (n_in < 120) return;  // expect jax pytree leaf order (see analysis)

  const float* enc   = (const float*)d_in[0];   // (B,SEQ,D)
  const float* times = (const float*)d_in[1];   // (B,SEQ,1)
  const float* x0    = (const float*)d_in[2];   // (B,29)
  // emb (sorted keys: Wb,Wc,Wp,Wt,bb,bc,bp,bt)
  const float* Wb = (const float*)d_in[3];
  const float* Wc = (const float*)d_in[4];
  const float* Wp = (const float*)d_in[5];
  const float* Wt = (const float*)d_in[6];
  const float* bb = (const float*)d_in[7];
  const float* bc = (const float*)d_in[8];
  const float* bp = (const float*)d_in[9];
  const float* bt = (const float*)d_in[10];
  // haz (W1,W2,b1,b2)
  const float* hazW1 = (const float*)d_in[11];
  const float* hazW2 = (const float*)d_in[12];
  const float* hazb1 = (const float*)d_in[13];
  const float* hazb2 = (const float*)d_in[14];
  // layers: keys W1,W2,b1,b2,ca{Wk,Wo,Wq,Wv,bk,bo,bq,bv},ln1b,ln1g,ln2b,ln2g,ln3b,ln3g,sa{...}
  const float *W1[4], *W2[4], *b1[4], *b2[4];
  const float *caWk[4], *caWo[4], *caWq[4], *caWv[4];
  const float *ca_bk[4], *ca_bo[4], *ca_bq[4], *ca_bv[4];
  const float *ln1b[4], *ln1g[4], *ln2b[4], *ln2g[4], *ln3b[4], *ln3g[4];
  const float *saWo[4], *saWv[4], *sa_bo[4], *sa_bv[4];
  for (int l = 0; l < 4; l++) {
    int base = 15 + 26 * l;
    W1[l]   = (const float*)d_in[base + 0];
    W2[l]   = (const float*)d_in[base + 1];
    b1[l]   = (const float*)d_in[base + 2];
    b2[l]   = (const float*)d_in[base + 3];
    caWk[l] = (const float*)d_in[base + 4];
    caWo[l] = (const float*)d_in[base + 5];
    caWq[l] = (const float*)d_in[base + 6];
    caWv[l] = (const float*)d_in[base + 7];
    ca_bk[l] = (const float*)d_in[base + 8];
    ca_bo[l] = (const float*)d_in[base + 9];
    ca_bq[l] = (const float*)d_in[base + 10];
    ca_bv[l] = (const float*)d_in[base + 11];
    ln1b[l] = (const float*)d_in[base + 12];
    ln1g[l] = (const float*)d_in[base + 13];
    ln2b[l] = (const float*)d_in[base + 14];
    ln2g[l] = (const float*)d_in[base + 15];
    ln3b[l] = (const float*)d_in[base + 16];
    ln3g[l] = (const float*)d_in[base + 17];
    saWo[l]  = (const float*)d_in[base + 19];
    saWv[l]  = (const float*)d_in[base + 21];
    sa_bo[l] = (const float*)d_in[base + 23];
    sa_bv[l] = (const float*)d_in[base + 25];
  }
  const float* pe = (const float*)d_in[119];    // (5000, D)

  // ---- workspace bump allocator (256B aligned regions) ----
  char* wsp = (char*)d_ws;
  size_t off = 0;
  auto alloc = [&](size_t bytes) -> void* {
    void* p = wsp + off;
    off = (off + bytes + 255) & ~(size_t)255;
    return p;
  };
  auto allocH = [&](size_t elems) -> _Float16* { return (_Float16*)alloc(elems * 2); };
  auto allocF = [&](size_t elems) -> float*    { return (float*)alloc(elems * 4); };

  _Float16* wcT    = allocH((size_t)D_ * 3 * D_);      // [512][1536]
  _Float16* hazW1T = allocH((size_t)HI_ * D_);         // [497][512]
  _Float16* encH   = allocH((size_t)B_ * SEQ_ * D_);   // f16 copy of enc
  _Float16 *saWvT[4], *saWoT[4], *caWqT[4], *caWkT[4], *caWvT[4], *caWoT[4], *w1T[4], *w2T[4];
  _Float16 *Kh[4], *Vh[4];
  for (int l = 0; l < 4; l++) {
    saWvT[l] = allocH((size_t)D_ * D_);
    saWoT[l] = allocH((size_t)D_ * D_);
    caWqT[l] = allocH((size_t)D_ * D_);
    caWkT[l] = allocH((size_t)D_ * D_);
    caWvT[l] = allocH((size_t)D_ * D_);
    caWoT[l] = allocH((size_t)D_ * D_);
    w1T[l]   = allocH((size_t)FF_ * D_);               // [2048][512]
    w2T[l]   = allocH((size_t)D_ * FF_);               // [512][2048]
    Kh[l]    = allocH((size_t)B_ * SEQ_ * D_);
    Vh[l]    = allocH((size_t)B_ * SEQ_ * D_);
  }
  float* e    = allocF((size_t)B_ * 3 * D_);
  float* xb   = allocF((size_t)B_ * D_);
  float* t1   = allocF((size_t)B_ * D_);
  float* t2   = allocF((size_t)B_ * D_);
  float* t3   = allocF((size_t)B_ * D_);
  float* ff   = allocF((size_t)B_ * FF_);
  float* hb   = allocF((size_t)B_ * D_);
  float* prev = allocF((size_t)B_ * 2);
  if (off > ws_size) return;  // insufficient scratch

  auto conv = [&](const float* W, _Float16* WT, int K, int N) {
    int tot = K * N;
    convT_kernel<<<(tot + 255) / 256, 256, 0, stream>>>(W, WT, K, N);
  };
  // per-step GEMM: M=512, fp32 A, 16x64 wave tile (occupancy for latency chain)
  auto gemmF = [&](const float* A, const _Float16* WT, const float* bias,
                   const float* extra, float* Cf, int N, int K, int epi) {
    int waves = (B_ >> 4) * ((N + 63) >> 6);
    int blocks = (waves + 7) >> 3;
    if (epi == 1)
      gemm_wmma_kernel<1, 0, 0, 1, 0><<<blocks, 256, 0, stream>>>(A, nullptr, WT, bias, extra, Cf, nullptr, B_, N, K);
    else if (epi == 2)
      gemm_wmma_kernel<2, 0, 0, 0, 0><<<blocks, 256, 0, stream>>>(A, nullptr, WT, bias, extra, Cf, nullptr, B_, N, K);
    else
      gemm_wmma_kernel<0, 0, 0, 1, 0><<<blocks, 256, 0, stream>>>(A, nullptr, WT, bias, extra, Cf, nullptr, B_, N, K);
  };
  // KV-precompute GEMM: M=25600, f16 A, 32x64 wave tile, f16 output
  auto gemmKV = [&](const _Float16* WT, const float* bias, _Float16* Ch) {
    int waves = ((B_ * SEQ_) >> 5) * (D_ >> 6);   // 800 * 8 = 6400
    int blocks = (waves + 7) >> 3;
    gemm_wmma_kernel<0, 1, 1, 1, 1><<<blocks, 256, 0, stream>>>(
        nullptr, encH, WT, bias, nullptr, nullptr, Ch, B_ * SEQ_, D_, D_);
  };

  // ---- one-time (per launch): weight convert+transpose, enc f16 copy ----
  conv(Wc, wcT, 3 * D_, D_);
  conv(hazW1, hazW1T, D_, HI_);
  for (int l = 0; l < 4; l++) {
    conv(saWv[l], saWvT[l], D_, D_);
    conv(saWo[l], saWoT[l], D_, D_);
    conv(caWq[l], caWqT[l], D_, D_);
    conv(caWk[l], caWkT[l], D_, D_);
    conv(caWv[l], caWvT[l], D_, D_);
    conv(caWo[l], caWoT[l], D_, D_);
    conv(W1[l], w1T[l], D_, FF_);
    conv(W2[l], w2T[l], FF_, D_);
  }
  {
    int n = B_ * SEQ_ * D_;
    convA_kernel<<<(n + 255) / 256, 256, 0, stream>>>(enc, encH, n);
  }

  // ---- KV cache precompute: K/V = enc @ Wk/Wv + b (f16 in, f16 out) ----
  for (int l = 0; l < 4; l++) {
    gemmKV(caWkT[l], ca_bk[l], Kh[l]);
    gemmKV(caWvT[l], ca_bv[l], Vh[l]);
  }

  // ---- autoregressive loop over 50 steps ----
  zero_prev_kernel<<<4, 256, 0, stream>>>(prev);
  for (int t = 0; t < SEQ_; t++) {
    embed_kernel<<<(B_ * 3 * D_ + 255) / 256, 256, 0, stream>>>(
        x0, times, prev, Wb, bb, Wt, bt, Wp, bp, e, t);
    // x = e @ Wc + bc + pe[t]
    gemmF(e, wcT, bc, pe + (size_t)t * D_, xb, D_, 3 * D_, 0);
    for (int l = 0; l < 4; l++) {
      // self-attn with seq=1: softmax==1 -> o = (x@Wv+bv)@Wo+bo exactly
      gemmF(xb, saWvT[l], sa_bv[l], nullptr, t1, D_, D_, 0);
      gemmF(t1, saWoT[l], sa_bo[l], nullptr, t2, D_, D_, 0);
      ln_kernel<<<64, 256, 0, stream>>>(xb, t2, ln1g[l], ln1b[l], xb);
      // cross-attention
      gemmF(xb, caWqT[l], ca_bq[l], nullptr, t1, D_, D_, 0);
      attn_kernel<<<B_ * NH_ / 8, 256, 0, stream>>>(t1, Kh[l], Vh[l], t2);
      gemmF(t2, caWoT[l], ca_bo[l], nullptr, t3, D_, D_, 0);
      ln_kernel<<<64, 256, 0, stream>>>(xb, t3, ln2g[l], ln2b[l], xb);
      // FFN
      gemmF(xb, w1T[l], b1[l], nullptr, ff, FF_, D_, 1);   // GELU
      gemmF(ff, w2T[l], b2[l], nullptr, t2, D_, FF_, 0);
      ln_kernel<<<64, 256, 0, stream>>>(xb, t2, ln3g[l], ln3b[l], xb);
    }
    // hazard head
    gemmF(xb, hazW1T, hazb1, nullptr, hb, HI_, D_, 2);     // ELU
    haz2_kernel<<<64, 256, 0, stream>>>(hb, hazW2, hazb2, prev, (float*)d_out, t);
  }
}
